// Transformer_83339545412186
// MI455X (gfx1250) — matrix operand order repaired and checked by reading, not compiled, not run on hardware
//
#include <hip/hip_runtime.h>

typedef __bf16 bf16_t;
typedef __attribute__((ext_vector_type(16))) __bf16 v16bf;
typedef __attribute__((ext_vector_type(8)))  float  v8f;

constexpr int kB  = 4;
constexpr int kC  = 512;   // channels
constexpr int kH  = 8;     // heads
constexpr int kKC = 64;    // head dim
constexpr int kT  = 1024;  // time
constexpr int kL  = 6;     // layers
constexpr int kK  = 3;     // conv kernel size
constexpr int kW  = 4;     // attention window
constexpr int kNR = 2 * kW + 1;  // 9 relative positions

union FragU { v16bf v; unsigned u[8]; };
union AccU  { v8f v; float f[8]; };

// ---- WMMA fragment loaders (CDNA5 ISA 7.12.2 layouts) ----
// A: 16x32 bf16 tile, element (m,k) at src[m*ld + k]; K pairs contiguous.
// lane: m = lane%16, hi = lane/16; reg j: k = 16*(j>=4) + 8*hi + 2*(j%4)
__device__ __forceinline__ v16bf frag_a(const bf16_t* src, int ld) {
  int lane = threadIdx.x & 31;
  int m = lane & 15, hi = lane >> 4;
  FragU f;
#pragma unroll
  for (int j = 0; j < 8; ++j) {
    int k = ((j & 4) << 2) + (hi << 3) + ((j & 3) << 1);
    f.u[j] = *reinterpret_cast<const unsigned*>(src + (size_t)m * ld + k);
  }
  return f.v;
}

// B: 32x16 bf16 tile, element (k,n) at src[n*ld + k] (B stored column-per-row).
// lane: n = lane%16, hi = lane/16; reg j: k = 16*(j>=4) + 8*hi + 2*(j%4)
__device__ __forceinline__ v16bf frag_b(const bf16_t* src, int ld) {
  int lane = threadIdx.x & 31;
  int n = lane & 15, hi = lane >> 4;
  FragU f;
#pragma unroll
  for (int j = 0; j < 8; ++j) {
    int k = ((j & 4) << 2) + (hi << 3) + ((j & 3) << 1);
    f.u[j] = *reinterpret_cast<const unsigned*>(src + (size_t)n * ld + k);
  }
  return f.v;
}

__device__ __forceinline__ v8f wmma_bf16(v16bf a, v16bf b, v8f c) {
  return __builtin_amdgcn_wmma_f32_16x16x32_bf16(false, a, false, b, (short)0, c,
                                                 false, false);
}

// 64x64 macro-tile: acc[i][j] += A(64 x Kdim)[16i..] * B(Kdim x 64)[.., 16j..]
// A rows at A + (16i)*lda; B cols from Bm + (16j)*ldb ([n][k] storage).
__device__ __forceinline__ void gemm_macro(const bf16_t* __restrict__ A, int lda,
                                           const bf16_t* __restrict__ Bm, int ldb,
                                           int Kdim, AccU acc[4][4]) {
  for (int k0 = 0; k0 < Kdim; k0 += 32) {
    v16bf a[4], b[4];
#pragma unroll
    for (int i = 0; i < 4; ++i) a[i] = frag_a(A + (size_t)(16 * i) * lda + k0, lda);
#pragma unroll
    for (int j = 0; j < 4; ++j) b[j] = frag_b(Bm + (size_t)(16 * j) * ldb + k0, ldb);
    if (k0 + 32 < Kdim) {  // prefetch next K-step (global_prefetch_b8 on gfx1250)
      int lane = threadIdx.x & 31;
      __builtin_prefetch(A + (size_t)(lane & 15) * lda + k0 + 32, 0, 3);
      __builtin_prefetch(Bm + (size_t)(lane & 15) * ldb + k0 + 32, 0, 3);
    }
#pragma unroll
    for (int i = 0; i < 4; ++i)
#pragma unroll
      for (int j = 0; j < 4; ++j)
        acc[i][j].v = wmma_bf16(a[i], b[j], acc[i][j].v);
  }
}

__device__ __forceinline__ void acc_zero(AccU acc[4][4]) {
#pragma unroll
  for (int i = 0; i < 4; ++i)
#pragma unroll
    for (int j = 0; j < 4; ++j)
#pragma unroll
      for (int r = 0; r < 8; ++r) acc[i][j].f[r] = 0.0f;
}

// ---- small utility kernels ----
__global__ void k_cvt_bf16(const float* __restrict__ src, bf16_t* __restrict__ dst,
                           size_t n) {
  size_t i = (size_t)blockIdx.x * blockDim.x + threadIdx.x;
  if (i < n) dst[i] = (bf16_t)src[i];
}

__global__ void k_init_x(const float* __restrict__ xin, float* __restrict__ xw,
                         bf16_t* __restrict__ xT) {
  size_t total = (size_t)kB * kC * kT;
  size_t i = (size_t)blockIdx.x * blockDim.x + threadIdx.x;
  if (i >= total) return;
  int t = (int)(i % kT);
  size_t r = i / kT;
  int c = (int)(r % kC);
  int b = (int)(r / kC);
  float v = xin[i];
  xw[i] = v;
  xT[((size_t)b * kT + t) * kC + c] = (bf16_t)v;
}

// im2col for K=3 'same' conv: dst[b][t][c*3+kk] = mask* (relu?) src[b][c][t+kk-1]
__global__ void k_im2col(const float* __restrict__ src, const int* __restrict__ lengths,
                         bf16_t* __restrict__ dst, int do_relu) {
  size_t total = (size_t)kB * kT * kC * kK;
  size_t i = (size_t)blockIdx.x * blockDim.x + threadIdx.x;
  if (i >= total) return;
  int kk = (int)(i % kK);
  size_t j = i / kK;
  int c = (int)(j % kC);
  size_t j2 = j / kC;
  int t = (int)(j2 % kT);
  int b = (int)(j2 / kT);
  int tt = t + kk - 1;
  float v = 0.0f;
  if (tt >= 0 && tt < kT && tt < lengths[b]) {
    v = src[((size_t)b * kC + c) * kT + tt];
    if (do_relu) v = fmaxf(v, 0.0f);
  }
  dst[i] = (bf16_t)v;
}

// ---- QKV projection: q/k/v = W * x + b; 64x64 tile per wave ----
// grid (kB*3, kC/64, kT/64), block 32
__global__ __launch_bounds__(32) void k_gemm_qkv(
    const bf16_t* __restrict__ Wq, const bf16_t* __restrict__ Wk,
    const bf16_t* __restrict__ Wv, const float* __restrict__ bq,
    const float* __restrict__ bk, const float* __restrict__ bv,
    const bf16_t* __restrict__ xT, bf16_t* __restrict__ qT,
    bf16_t* __restrict__ kTr, bf16_t* __restrict__ vB) {
  int which = blockIdx.x % 3;
  int b = blockIdx.x / 3;
  int m0 = blockIdx.y * 64;
  int n0 = blockIdx.z * 64;
  const bf16_t* W = (which == 0) ? Wq : (which == 1) ? Wk : Wv;
  const float* bias = (which == 0) ? bq : (which == 1) ? bk : bv;
  AccU acc[4][4];
  acc_zero(acc);
  gemm_macro(W + (size_t)m0 * kC, kC,
             xT + ((size_t)b * kT + n0) * kC, kC, kC, acc);
  int lane = threadIdx.x & 31, n = lane & 15, hi = lane >> 4;
#pragma unroll
  for (int i = 0; i < 4; ++i)
#pragma unroll
    for (int j = 0; j < 4; ++j)
#pragma unroll
      for (int r = 0; r < 8; ++r) {
        int m = m0 + 16 * i + r + 8 * hi;
        int t = n0 + 16 * j + n;
        float v = acc[i][j].f[r] + bias[m];
        if (which == 0)
          qT[((size_t)b * kT + t) * kC + m] = (bf16_t)(v * 0.125f);  // pre-scale q
        else if (which == 1)
          kTr[((size_t)b * kT + t) * kC + m] = (bf16_t)v;
        else
          vB[((size_t)b * kC + m) * kT + t] = (bf16_t)v;
      }
}

// ---- generic GEMM + epilogue: out[b][m][t]; 64x64 tile per wave ----
// mode 0: out = acc + bias + xres            (attention out-proj + residual)
// mode 1: out = acc + bias                   (ffn hidden)
// mode 2: out = xres + mask[t]*(acc + bias)  (ffn out, masked, + residual)
__global__ __launch_bounds__(32) void k_gemm_ep(
    const bf16_t* __restrict__ A, const bf16_t* __restrict__ Bm, int Kdim,
    const float* __restrict__ bias, const float* __restrict__ xres,
    float* __restrict__ out, int mode, const int* __restrict__ lengths) {
  int b = blockIdx.x, m0 = blockIdx.y * 64, n0 = blockIdx.z * 64;
  AccU acc[4][4];
  acc_zero(acc);
  gemm_macro(A + (size_t)m0 * Kdim, Kdim,
             Bm + ((size_t)b * kT + n0) * (size_t)Kdim, Kdim, Kdim, acc);
  int lane = threadIdx.x & 31, n = lane & 15, hi = lane >> 4;
  int len = lengths[b];
#pragma unroll
  for (int i = 0; i < 4; ++i)
#pragma unroll
    for (int j = 0; j < 4; ++j)
#pragma unroll
      for (int r = 0; r < 8; ++r) {
        int m = m0 + 16 * i + r + 8 * hi;
        int t = n0 + 16 * j + n;
        float v = acc[i][j].f[r] + bias[m];
        if (mode == 2 && t >= len) v = 0.0f;
        if (mode != 1) v += xres[((size_t)b * kC + m) * kT + t];
        out[((size_t)b * kC + m) * kT + t] = v;
      }
}

// ---- fused windowed-relative attention, flash-style; 1 wave per (b,h,t-tile) ----
__global__ __launch_bounds__(32) void k_attn(
    const bf16_t* __restrict__ qT, const bf16_t* __restrict__ kTr,
    const bf16_t* __restrict__ vB, const float* __restrict__ erk,
    const float* __restrict__ erv, const int* __restrict__ lengths,
    bf16_t* __restrict__ attnT) {
  int bh = blockIdx.x;
  int b = bh / kH, h = bh % kH;
  int t0 = blockIdx.y * 16;
  int lane = threadIdx.x & 31;
  int n = lane & 15, hi = lane >> 4;
  int len = lengths[b];

  __shared__ float s_rq[16 * kNR];     // (q*scale) . erk[delta]
  __shared__ float s_erv[kNR * kKC];   // erv cached
  __shared__ bf16_t s_p[16 * 32];      // P tile staging (C-layout -> A-layout)

  for (int i = lane; i < kNR * kKC; i += 32) s_erv[i] = erv[i];

  const bf16_t* qbase = qT + ((size_t)b * kT + t0) * kC + h * kKC;
  for (int i = lane; i < 16 * kNR; i += 32) {
    int row = i / kNR, dd = i % kNR;
    float s = 0.0f;
    for (int c = 0; c < kKC; ++c)
      s += (float)qbase[(size_t)row * kC + c] * erk[dd * kKC + c];
    s_rq[i] = s;
  }
  __syncthreads();

  v16bf qa0 = frag_a(qbase, kC);
  v16bf qa1 = frag_a(qbase + 32, kC);

  float rm[8], rl[8];
  AccU oacc[4];
#pragma unroll
  for (int r = 0; r < 8; ++r) { rm[r] = -1e30f; rl[r] = 0.0f; }
#pragma unroll
  for (int dc = 0; dc < 4; ++dc)
#pragma unroll
    for (int r = 0; r < 8; ++r) oacc[dc].f[r] = 0.0f;

  for (int s0 = 0; s0 < kT; s0 += 32) {
    AccU S[2];
#pragma unroll
    for (int half = 0; half < 2; ++half) {
#pragma unroll
      for (int r = 0; r < 8; ++r) S[half].f[r] = 0.0f;
      const bf16_t* kb = kTr + ((size_t)b * kT + s0 + half * 16) * kC + h * kKC;
      v16bf b0 = frag_b(kb, kC);
      S[half].v = wmma_bf16(qa0, b0, S[half].v);
      v16bf b1 = frag_b(kb + 32, kC);
      S[half].v = wmma_bf16(qa1, b1, S[half].v);
    }
    // relative logits (banded) + mask
#pragma unroll
    for (int half = 0; half < 2; ++half) {
#pragma unroll
      for (int r = 0; r < 8; ++r) {
        int m = r + 8 * hi;
        int tg = t0 + m;
        int sg = s0 + half * 16 + n;
        float v = S[half].f[r];
        int d = sg - tg;
        if (d >= -kW && d <= kW) v += s_rq[m * kNR + d + kW];
        if (!(tg < len && sg < len)) v = -10000.0f;
        S[half].f[r] = v;
      }
    }
    __syncthreads();  // prior s_p readers done before overwrite
    // online softmax per row (row r+8*hi lives across lanes of same half)
#pragma unroll
    for (int r = 0; r < 8; ++r) {
      int m = r + 8 * hi;
      float mx = fmaxf(S[0].f[r], S[1].f[r]);
#pragma unroll
      for (int off = 1; off < 16; off <<= 1) mx = fmaxf(mx, __shfl_xor(mx, off, 32));
      float nm = fmaxf(rm[r], mx);
      float e0 = __expf(S[0].f[r] - nm);
      float e1 = __expf(S[1].f[r] - nm);
      float ps = e0 + e1;
#pragma unroll
      for (int off = 1; off < 16; off <<= 1) ps += __shfl_xor(ps, off, 32);
      float corr = __expf(rm[r] - nm);
      rl[r] = rl[r] * corr + ps;
      rm[r] = nm;
#pragma unroll
      for (int dc = 0; dc < 4; ++dc) oacc[dc].f[r] *= corr;
      s_p[m * 32 + n] = (bf16_t)e0;
      s_p[m * 32 + 16 + n] = (bf16_t)e1;
    }
    __syncthreads();
    v16bf pa = frag_a(s_p, 32);
#pragma unroll
    for (int dc = 0; dc < 4; ++dc) {
      const bf16_t* vb = vB + ((size_t)b * kC + h * kKC + dc * 16) * kT + s0;
      v16bf bvf = frag_b(vb, kT);
      oacc[dc].v = wmma_bf16(pa, bvf, oacc[dc].v);
    }
    // banded attn . erv contribution (unnormalized; fixed by later corr & 1/l)
    if (s0 <= t0 + 19 && s0 + 31 >= t0 - kW) {
#pragma unroll
      for (int r = 0; r < 8; ++r) {
        int m = r + 8 * hi;
        int tg = t0 + m;
#pragma unroll
        for (int dd = 0; dd < kNR; ++dd) {
          int sg = tg + dd - kW;
          if (sg >= s0 && sg < s0 + 32) {
            float pval = (float)s_p[m * 32 + (sg - s0)];
#pragma unroll
            for (int dc = 0; dc < 4; ++dc)
              oacc[dc].f[r] += pval * s_erv[dd * kKC + dc * 16 + n];
          }
        }
      }
    }
  }
#pragma unroll
  for (int r = 0; r < 8; ++r) {
    int m = r + 8 * hi;
    int tg = t0 + m;
    float inv = 1.0f / rl[r];
#pragma unroll
    for (int dc = 0; dc < 4; ++dc)
      attnT[((size_t)b * kT + tg) * kC + h * kKC + dc * 16 + n] =
          (bf16_t)(oacc[dc].f[r] * inv);
  }
}

// ---- per-channel LayerNorm over C; writes fp32 x and bf16 xT; optional relu ----
__global__ __launch_bounds__(32) void k_layernorm(const float* __restrict__ z,
                                                  float* __restrict__ xout,
                                                  bf16_t* __restrict__ xT,
                                                  int do_relu) {
  int b = blockIdx.x, t = blockIdx.y, lane = threadIdx.x & 31;
  float vals[16];
  float s = 0.0f;
#pragma unroll
  for (int i = 0; i < 16; ++i) {
    int c = lane + i * 32;
    vals[i] = z[((size_t)b * kC + c) * kT + t];
    s += vals[i];
  }
#pragma unroll
  for (int off = 1; off < 32; off <<= 1) s += __shfl_xor(s, off, 32);
  float mean = s * (1.0f / kC);
  float vs = 0.0f;
#pragma unroll
  for (int i = 0; i < 16; ++i) { float d = vals[i] - mean; vs += d * d; }
#pragma unroll
  for (int off = 1; off < 32; off <<= 1) vs += __shfl_xor(vs, off, 32);
  float inv = rsqrtf(vs * (1.0f / kC) + 1e-5f);
#pragma unroll
  for (int i = 0; i < 16; ++i) {
    int c = lane + i * 32;
    float o = (vals[i] - mean) * inv;
    if (do_relu) o = fmaxf(o, 0.0f);
    xout[((size_t)b * kC + c) * kT + t] = o;
    xT[((size_t)b * kT + t) * kC + c] = (bf16_t)o;
  }
}

extern "C" void kernel_launch(void* const* d_in, const int* in_sizes, int n_in,
                              void* d_out, int out_size, void* d_ws, size_t ws_size,
                              hipStream_t stream) {
  (void)in_sizes; (void)n_in; (void)out_size; (void)ws_size;
  const float* x_in = (const float*)d_in[0];
  const int* lengths = (const int*)d_in[1];
  const float* Wq = (const float*)d_in[2];
  const float* bq = (const float*)d_in[3];
  const float* Wk = (const float*)d_in[4];
  const float* bk = (const float*)d_in[5];
  const float* Wv = (const float*)d_in[6];
  const float* bv = (const float*)d_in[7];
  const float* Wo = (const float*)d_in[8];
  const float* bo = (const float*)d_in[9];
  const float* erk = (const float*)d_in[10];
  const float* erv = (const float*)d_in[11];
  const float* Wf1 = (const float*)d_in[12];
  const float* bf1 = (const float*)d_in[13];
  const float* Wf2 = (const float*)d_in[14];
  const float* bf2 = (const float*)d_in[15];

  char* p = (char*)d_ws;
  auto carve = [&](size_t bytes) {
    void* r = (void*)p;
    p += (bytes + 255) & ~(size_t)255;
    return r;
  };
  const size_t WCC = (size_t)kL * kC * kC;
  const size_t NBCT = (size_t)kB * kC * kT;
  bf16_t* Wq_b = (bf16_t*)carve(WCC * 2);
  bf16_t* Wk_b = (bf16_t*)carve(WCC * 2);
  bf16_t* Wv_b = (bf16_t*)carve(WCC * 2);
  bf16_t* Wo_b = (bf16_t*)carve(WCC * 2);
  bf16_t* W1_b = (bf16_t*)carve(WCC * kK * 2);
  bf16_t* W2_b = (bf16_t*)carve(WCC * kK * 2);
  float* xw = (float*)carve(NBCT * 4);
  float* tmp = (float*)carve(NBCT * 4);
  float* hbuf = (float*)carve(NBCT * 4);
  bf16_t* xT = (bf16_t*)carve(NBCT * 2);
  bf16_t* qT = (bf16_t*)carve(NBCT * 2);
  bf16_t* kTb = (bf16_t*)carve(NBCT * 2);
  bf16_t* vBb = (bf16_t*)carve(NBCT * 2);
  bf16_t* aT = (bf16_t*)carve(NBCT * 2);
  bf16_t* xcol = (bf16_t*)carve(NBCT * kK * 2);

  auto cvt = [&](const float* s, bf16_t* d, size_t nn) {
    k_cvt_bf16<<<(unsigned)((nn + 255) / 256), 256, 0, stream>>>(s, d, nn);
  };
  cvt(Wq, Wq_b, WCC);
  cvt(Wk, Wk_b, WCC);
  cvt(Wv, Wv_b, WCC);
  cvt(Wo, Wo_b, WCC);
  cvt(Wf1, W1_b, WCC * kK);
  cvt(Wf2, W2_b, WCC * kK);

  k_init_x<<<(unsigned)((NBCT + 255) / 256), 256, 0, stream>>>(x_in, xw, xT);

  dim3 gGemm(kB, kC / 64, kT / 64);
  size_t colN = NBCT * kK;
  for (int l = 0; l < kL; ++l) {
    const bf16_t* wq = Wq_b + (size_t)l * kC * kC;
    const bf16_t* wk = Wk_b + (size_t)l * kC * kC;
    const bf16_t* wv = Wv_b + (size_t)l * kC * kC;
    const bf16_t* wo = Wo_b + (size_t)l * kC * kC;
    const bf16_t* w1 = W1_b + (size_t)l * kC * kC * kK;
    const bf16_t* w2 = W2_b + (size_t)l * kC * kC * kK;

    k_gemm_qkv<<<dim3(kB * 3, kC / 64, kT / 64), 32, 0, stream>>>(
        wq, wk, wv, bq + l * kC, bk + l * kC, bv + l * kC, xT, qT, kTb, vBb);
    k_attn<<<dim3(kB * kH, kT / 16), 32, 0, stream>>>(
        qT, kTb, vBb, erk + (size_t)l * kNR * kKC, erv + (size_t)l * kNR * kKC,
        lengths, aT);
    k_gemm_ep<<<gGemm, 32, 0, stream>>>(wo, aT, kC, bo + l * kC, xw, tmp, 0,
                                        lengths);
    k_layernorm<<<dim3(kB, kT), 32, 0, stream>>>(tmp, xw, xT, 0);

    k_im2col<<<(unsigned)((colN + 255) / 256), 256, 0, stream>>>(xw, lengths,
                                                                 xcol, 0);
    k_gemm_ep<<<gGemm, 32, 0, stream>>>(w1, xcol, kC * kK, bf1 + l * kC, xw,
                                        hbuf, 1, lengths);
    k_im2col<<<(unsigned)((colN + 255) / 256), 256, 0, stream>>>(hbuf, lengths,
                                                                 xcol, 1);
    k_gemm_ep<<<gGemm, 32, 0, stream>>>(w2, xcol, kC * kK, bf2 + l * kC, xw,
                                        tmp, 2, lengths);
    int last = (l == kL - 1);
    k_layernorm<<<dim3(kB, kT), 32, 0, stream>>>(
        tmp, last ? (float*)d_out : xw, xT, last ? 0 : 1);
  }
}
